// MutualCrossAttention_111669149784
// MI455X (gfx1250) — matrix operand order repaired and checked by reading, not compiled, not run on hardware
//
#include <hip/hip_runtime.h>
#include <hip/hip_bf16.h>
#include <cstdint>

// ---------------------------------------------------------------------------
// MutualCrossAttention fused kernel for gfx1250 (MI455X, wave32, WMMA).
//
// Shapes: T=4, B=16, Nq=2048, Nkv=4, D=256, H=8, d=32.  TAU=2, VTH=0.5.
// Bandwidth-bound (~268MB traffic vs ~35 GFLOP) -> fully fused, f16 WMMA
// for the two 256x256 GEMMs, everything else in registers/LDS.
// ---------------------------------------------------------------------------

typedef __attribute__((ext_vector_type(16))) _Float16 v16h;
typedef __attribute__((ext_vector_type(8)))  _Float16 v8h;
typedef __attribute__((ext_vector_type(8)))  float    v8f;

union AF {
    v16h v;
    struct { v8h lo, hi; } p;
};

__device__ __forceinline__ v8f wmma_f16(v16h a, v16h b, v8f c) {
    // D = A(16x32 f16) * B(32x16 f16) + C(16x16 f32)
    return __builtin_amdgcn_wmma_f32_16x16x32_f16(
        /*neg_a=*/false, a, /*neg_b=*/false, b,
        /*c_mod=*/(short)0, c, /*reuse_a=*/false, /*reuse_b=*/false);
}

// ---------------------------------------------------------------------------
// Setup kernel 1: convert Wq / Wp (256x256 f32) to f16 in workspace.
// ---------------------------------------------------------------------------
__global__ void setup_w(const float* __restrict__ Wq, const float* __restrict__ Wp,
                        _Float16* __restrict__ Wq16, _Float16* __restrict__ Wp16) {
    int i = blockIdx.x * 256 + threadIdx.x;   // grid 256 x 256 threads = 65536
    Wq16[i] = (_Float16)Wq[i];
    Wp16[i] = (_Float16)Wp[i];
}

// ---------------------------------------------------------------------------
// Setup kernel 2: K/V paths on the tiny kv tensor.
//   Ksc[b][tok][j][e] = BN_k( kv[j,b,tok,:] @ Wk^T )[e]
//   Vsc[b][tok][j][e] = LIF-over-j( BN_v( kv[j,b,tok,:] @ Wv^T ) )  (binary)
// (tok plays the role of the q time step t after the reference's swapaxes.)
// ---------------------------------------------------------------------------
__global__ __launch_bounds__(256) void setup_kv(
    const float* __restrict__ kv, const float* __restrict__ Wk,
    const float* __restrict__ Wv, const float* __restrict__ bnk,
    const float* __restrict__ bnv, float* __restrict__ Ksc,
    float* __restrict__ Vsc) {
    const int bt  = blockIdx.x;          // 0..63 : b*4 + tok
    const int b   = bt >> 2;
    const int tok = bt & 3;
    const int e   = threadIdx.x;         // channel 0..255

    __shared__ float row[4][256];
    #pragma unroll
    for (int j = 0; j < 4; ++j)
        row[j][e] = kv[((j * 16 + b) * 4 + tok) * 256 + e];
    __syncthreads();

    const float kscale = bnk[e] * rsqrtf(bnk[768 + e] + 1e-5f);
    const float kshift = bnk[256 + e] - bnk[512 + e] * kscale;
    const float vscale = bnv[e] * rsqrtf(bnv[768 + e] + 1e-5f);
    const float vshift = bnv[256 + e] - bnv[512 + e] * vscale;

    const float* wkr = Wk + e * 256;
    const float* wvr = Wv + e * 256;

    float vstate = 0.f;
    for (int j = 0; j < 4; ++j) {
        float accK = 0.f, accV = 0.f;
        #pragma unroll 8
        for (int d0 = 0; d0 < 256; ++d0) {
            float x = row[j][d0];
            accK += x * wkr[d0];
            accV += x * wvr[d0];
        }
        Ksc[(bt * 4 + j) * 256 + e] = accK * kscale + kshift;
        // LIF (tau=2, vth=0.5, hard reset)
        float xv = accV * vscale + vshift;
        float h  = vstate + (xv - vstate) * 0.5f;
        float s  = (h >= 0.5f) ? 1.f : 0.f;
        vstate   = h * (1.f - s);
        Vsc[(bt * 4 + j) * 256 + e] = s;
    }
}

// ---------------------------------------------------------------------------
// GEMM helper: 64x256 += A(64x256 f16, LDS) * Bg^T-style(256x256 f16, global).
// Each wave owns 32 output channels (2 N-tiles), all 4 M-tiles.
// Fragment addressing per CDNA5 ISA 7.12.2 (wave32):
//   A 16x32:  lane<16 -> K {0..7,16..23}, lane>=16 -> K {8..15,24..31}
//   B 32x16:  lane%16 = N, same K split as A
// ---------------------------------------------------------------------------
__device__ __forceinline__ void gemm64x256(const _Float16* __restrict__ A,
                                           const _Float16* __restrict__ Bg,
                                           v8f acc[4][2],
                                           int wave, int llo, int lhi) {
    #pragma unroll
    for (int k0 = 0; k0 < 8; ++k0) {
        AF a[4], w[2];
        #pragma unroll
        for (int mt = 0; mt < 4; ++mt) {
            const _Float16* ap = A + (mt * 16 + llo) * 256 + k0 * 32 + lhi * 8;
            a[mt].p.lo = *(const v8h*)(ap);        // ds_read_b128
            a[mt].p.hi = *(const v8h*)(ap + 16);
        }
        #pragma unroll
        for (int nt = 0; nt < 2; ++nt) {
            const _Float16* bp = Bg + (wave * 32 + nt * 16 + llo) * 256
                                    + k0 * 32 + lhi * 8;
            w[nt].p.lo = *(const v8h*)(bp);        // global_load_b128
            w[nt].p.hi = *(const v8h*)(bp + 16);
        }
        #pragma unroll
        for (int mt = 0; mt < 4; ++mt)
            #pragma unroll
            for (int nt = 0; nt < 2; ++nt)
                acc[mt][nt] = wmma_f16(a[mt].v, w[nt].v, acc[mt][nt]);
    }
}

// ---------------------------------------------------------------------------
// Main fused kernel. Grid (32, 16, 4) = (c-block, b, t); 256 threads = 8 waves.
// WG owns 64 q rows: m = 16*k + r  ->  query i = 512*k + cbase + r,
// so both post-attention LIF chains (over k) stay inside the WG.
// ---------------------------------------------------------------------------
__global__ __launch_bounds__(256) void mca_main(
    const float*    __restrict__ q,
    const _Float16* __restrict__ Wq16,
    const _Float16* __restrict__ Wp16,
    const float*    __restrict__ Ksc,
    const float*    __restrict__ Vsc,
    const float*    __restrict__ bnq,
    const float*    __restrict__ bnp,
    float*          __restrict__ out) {

    __shared__ __attribute__((aligned(16))) _Float16 sAB[64 * 256]; // q -> qh -> spikes
    __shared__ __attribute__((aligned(16))) _Float16 sK[16 * 256];  // K, rows 4..15 = 0
    __shared__ __attribute__((aligned(16))) float    sV[4 * 256];   // V spikes
    __shared__ __attribute__((aligned(16))) float    sSc[64 * 8 * 4]; // scores

    const int tid  = threadIdx.x;
    const int wave = tid >> 5;
    const int lane = tid & 31;
    const int lhi  = lane >> 4;   // half-wave select
    const int llo  = lane & 15;

    const int cblk  = blockIdx.x;       // 0..31
    const int b     = blockIdx.y;       // 0..15
    const int t     = blockIdx.z;       // 0..3
    const int cbase = cblk * 16;

    // ---- stage 0: stage q tile (f32 -> f16) and K/V into LDS --------------
    const float* qbase = q + ((size_t)(t * 16 + b) * 2048) * 256;
    #pragma unroll 4
    for (int m = 0; m < 64; ++m) {
        int i = ((m >> 4) << 9) + cbase + (m & 15);       // 512*k + c
        if (m + 4 < 64) {
            int inx = (((m + 4) >> 4) << 9) + cbase + ((m + 4) & 15);
            __builtin_prefetch(qbase + (size_t)inx * 256 + tid, 0, 1);
        }
        sAB[m * 256 + tid] = (_Float16)qbase[(size_t)i * 256 + tid];
    }
    const int btoff = (b * 4 + t) * 4 * 256;
    #pragma unroll
    for (int j = 0; j < 16; ++j)
        sK[j * 256 + tid] = (j < 4) ? (_Float16)Ksc[btoff + j * 256 + tid]
                                    : (_Float16)0.f;
    #pragma unroll
    for (int j = 0; j < 4; ++j)
        sV[j * 256 + tid] = Vsc[btoff + j * 256 + tid];

    // per-lane BN coefficients for its two channel slots
    float qsc[2], qsh[2], psc[2], psh[2];
    #pragma unroll
    for (int nt = 0; nt < 2; ++nt) {
        int ch = wave * 32 + nt * 16 + llo;
        float g = bnq[ch], be = bnq[256 + ch], mu = bnq[512 + ch], va = bnq[768 + ch];
        qsc[nt] = g * rsqrtf(va + 1e-5f);
        qsh[nt] = be - mu * qsc[nt];
        g = bnp[ch]; be = bnp[256 + ch]; mu = bnp[512 + ch]; va = bnp[768 + ch];
        psc[nt] = g * rsqrtf(va + 1e-5f);
        psh[nt] = be - mu * psc[nt];
    }
    __syncthreads();

    // ---- stage 1: qh = BN_q(q @ Wq^T), 64x256 via WMMA --------------------
    v8f acc[4][2] = {};
    gemm64x256(sAB, Wq16, acc, wave, llo, lhi);
    __syncthreads();                       // all reads of q tile complete
    #pragma unroll
    for (int mt = 0; mt < 4; ++mt)
        #pragma unroll
        for (int nt = 0; nt < 2; ++nt) {
            int ch = wave * 32 + nt * 16 + llo;
            #pragma unroll
            for (int r = 0; r < 8; ++r) {
                int m = mt * 16 + r + lhi * 8;     // C layout: M = r (+8 hi half)
                sAB[m * 256 + ch] = (_Float16)(acc[mt][nt][r] * qsc[nt] + qsh[nt]);
            }
        }
    __syncthreads();

    // ---- stage 2: per-head scores = qh_head @ K_head^T (K-dim 32) ---------
    {
        const int h = wave;                 // 8 waves <-> 8 heads
        AF bK;
        const _Float16* bp = sK + llo * 256 + h * 32 + lhi * 8;
        bK.p.lo = *(const v8h*)(bp);
        bK.p.hi = *(const v8h*)(bp + 16);
        #pragma unroll
        for (int mt = 0; mt < 4; ++mt) {
            AF a;
            const _Float16* ap = sAB + (mt * 16 + llo) * 256 + h * 32 + lhi * 8;
            a.p.lo = *(const v8h*)(ap);
            a.p.hi = *(const v8h*)(ap + 16);
            v8f sc = {};
            sc = wmma_f16(a.v, bK.v, sc);
            if (llo < 4) {
                #pragma unroll
                for (int r = 0; r < 8; ++r) {
                    int m = mt * 16 + r + lhi * 8;
                    sSc[(m * 8 + h) * 4 + llo] = sc[r];
                }
            }
        }
    }
    __syncthreads();

    // ---- stage 3: z = (scores @ V) * D^-0.5, LIF chain #1 -> spikes -------
    {
        const int ch = tid;                  // one channel per thread
        const int hh = ch >> 5;
        const float v0 = sV[0 * 256 + ch], v1 = sV[1 * 256 + ch];
        const float v2 = sV[2 * 256 + ch], v3 = sV[3 * 256 + ch];
        #pragma unroll 4
        for (int r = 0; r < 16; ++r) {
            float vs = 0.f;                  // LIF state along chain k
            #pragma unroll
            for (int k = 0; k < 4; ++k) {
                int m = k * 16 + r;
                const float* sp = sSc + (m * 8 + hh) * 4;
                float z = (sp[0] * v0 + sp[1] * v1 + sp[2] * v2 + sp[3] * v3)
                          * 0.0625f;         // scale = D^-0.5 = 1/16
                float hm = vs + (z - vs) * 0.5f;
                float s  = (hm >= 0.5f) ? 1.f : 0.f;
                vs = hm * (1.f - s);
                sAB[m * 256 + ch] = (_Float16)s;   // spikes overwrite qh
            }
        }
    }
    __syncthreads();

    // ---- stage 4: y = BN_p(spikes @ Wp^T); LIF chain #2; scattered store --
    v8f acc2[4][2] = {};
    gemm64x256(sAB, Wp16, acc2, wave, llo, lhi);

    // out layout (t2=4, b2=16, n2=2048, e=256):
    //   t2 = chain step k (= M-tile g), b2 = c>>5, n2 = (c&31)*64 + b*4 + t
    #pragma unroll
    for (int nt = 0; nt < 2; ++nt) {
        const int ch = wave * 32 + nt * 16 + llo;
        #pragma unroll
        for (int r = 0; r < 8; ++r) {
            const int c = cbase + r + lhi * 8;
            size_t baseaddr = (size_t)(c >> 5) * 524288
                            + (size_t)(((c & 31) * 64) + b * 4 + t) * 256 + ch;
            float vs = 0.f;
            #pragma unroll
            for (int g = 0; g < 4; ++g) {    // chain over M-tile accumulators
                float y  = acc2[g][nt][r] * psc[nt] + psh[nt];
                float hm = vs + (y - vs) * 0.5f;
                float s  = (hm >= 0.5f) ? 1.f : 0.f;
                vs = hm * (1.f - s);
                out[(size_t)g * 8388608 + baseaddr] = s;
            }
        }
    }
}

// ---------------------------------------------------------------------------
// Launch. Inputs: q, kv, Wq, Wk, Wv, Wp, bn_q, bn_k, bn_v, bn_p (all f32).
// ---------------------------------------------------------------------------
extern "C" void kernel_launch(void* const* d_in, const int* in_sizes, int n_in,
                              void* d_out, int out_size, void* d_ws, size_t ws_size,
                              hipStream_t stream) {
    (void)in_sizes; (void)n_in; (void)out_size; (void)ws_size;

    const float* q   = (const float*)d_in[0];
    const float* kv  = (const float*)d_in[1];
    const float* Wq  = (const float*)d_in[2];
    const float* Wk  = (const float*)d_in[3];
    const float* Wv  = (const float*)d_in[4];
    const float* Wp  = (const float*)d_in[5];
    const float* bnq = (const float*)d_in[6];
    const float* bnk = (const float*)d_in[7];
    const float* bnv = (const float*)d_in[8];
    const float* bnp = (const float*)d_in[9];
    float* out = (float*)d_out;

    char* ws = (char*)d_ws;
    _Float16* Wq16 = (_Float16*)(ws);                 // 128 KB
    _Float16* Wp16 = (_Float16*)(ws + (131072));      // 128 KB
    float*    Ksc  = (float*)   (ws + (262144));      // 256 KB
    float*    Vsc  = (float*)   (ws + (524288));      // 256 KB

    setup_w <<<256, 256, 0, stream>>>(Wq, Wp, Wq16, Wp16);
    setup_kv<<<64,  256, 0, stream>>>(kv, Wk, Wv, bnk, bnv, Ksc, Vsc);
    mca_main<<<dim3(32, 16, 4), 256, 0, stream>>>(q, Wq16, Wp16, Ksc, Vsc,
                                                  bnq, bnp, out);
}
